// TileWiseFloat8GroupedLinear_51230369907469
// MI455X (gfx1250) — 8-bit matrix operand order repaired and checked by reading, compiled, not run on hardware
//
#include <hip/hip_runtime.h>
#include <stdint.h>

typedef __attribute__((ext_vector_type(16))) int   v16i;
typedef __attribute__((ext_vector_type(8)))  float v8f;
typedef __attribute__((ext_vector_type(4)))  float v4f;

#define NE_   8
#define DIN_  2048
#define DOUT_ 1024
#define SEQ_  16384
#define TPE_  (SEQ_/NE_)
#define KB_   (DIN_/128)
#define NT_   (DOUT_/16)
#define OB_   (DOUT_/128)
#define RTE_  (TPE_/16)
#define FRAG_DW 512

__device__ __forceinline__ void async_copy_b128(const uint32_t* gsrc, void* ldst) {
  uint32_t loff = (uint32_t)(uintptr_t)ldst;
  asm volatile("global_load_async_to_lds_b128 %0, %1, off"
               :: "v"(loff), "v"((uint64_t)(uintptr_t)gsrc)
               : "memory");
}

__device__ __forceinline__ void wait_async0() {
#if __has_builtin(__builtin_amdgcn_s_wait_asynccnt)
  __builtin_amdgcn_s_wait_asynccnt(0);
#else
  asm volatile("s_wait_asynccnt 0x0" ::: "memory");
#endif
}

__device__ __forceinline__ uint32_t f32_to_e4m3(float x) {
  uint32_t bits = __float_as_uint(x);
  uint32_t sign = (bits >> 24) & 0x80u;
  float af = fminf(fabsf(x), 448.0f);
  uint32_t q;
  if (af < 0.015625f) {
    int mi = (int)rintf(af * 512.0f);
    q = (mi >= 8) ? 0x08u : (uint32_t)mi;
  } else {
    uint32_t b   = __float_as_uint(af);
    int      e   = (int)(b >> 23) - 127;
    uint32_t man = (b >> 20) & 0x7u;
    uint32_t rem = b & 0xFFFFFu;
    uint32_t E   = (uint32_t)(e + 7);
    if (rem > 0x80000u || (rem == 0x80000u && (man & 1u))) {
      man++;
      if (man == 8u) { man = 0u; E++; }
    }
    q = (E << 3) | man;
  }
  return q | sign;
}

__device__ __forceinline__ uint32_t pack4_e4m3(float a, float b, float c, float d) {
  return f32_to_e4m3(a) | (f32_to_e4m3(b) << 8) |
         (f32_to_e4m3(c) << 16) | (f32_to_e4m3(d) << 24);
}

__global__ void __launch_bounds__(256) quant_act(const float* __restrict__ x,
                                                 uint32_t* __restrict__ qxp,
                                                 float* __restrict__ sx) {
  int warp = threadIdx.x >> 5;
  int lane = threadIdx.x & 31;
  int tile = blockIdx.x * 8 + warp;
  int s  = tile / KB_;
  int kb = tile - s * KB_;

  const v4f v = __builtin_nontemporal_load(
      (const v4f*)(x + (size_t)s * DIN_ + kb * 128 + lane * 4));
  float amax = fmaxf(fmaxf(fabsf(v[0]), fabsf(v[1])), fmaxf(fabsf(v[2]), fabsf(v[3])));
  #pragma unroll
  for (int off = 16; off >= 1; off >>= 1)
    amax = fmaxf(amax, __shfl_xor(amax, off, 32));

  float scale = fmaxf(amax, 1e-12f) * (1.0f / 448.0f);
  float inv   = 1.0f / scale;
  uint32_t d  = pack4_e4m3(v[0] * inv, v[1] * inv, v[2] * inv, v[3] * inv);

  int kp = lane * 4;
  int h  = kp >> 6;
  int g  = (kp >> 3) & 1;
  int V  = h * 8 + ((kp & 63) >> 4) * 2 + ((kp >> 2) & 1);
  int rt = s >> 4, M = s & 15;
  int L  = M + 16 * g;
  qxp[(((size_t)rt * KB_ + kb) * 32 + L) * 16 + V] = d;
  if (lane == 0) sx[(size_t)s * KB_ + kb] = scale;
}

__global__ void __launch_bounds__(256) quant_wgt(const float* __restrict__ w,
                                                 uint32_t* __restrict__ wp,
                                                 float* __restrict__ sw) {
  __shared__ float red[256];
  int bid = blockIdx.x;
  int kb  = bid % KB_;
  int ob  = (bid / KB_) % OB_;
  int e   = bid / (KB_ * OB_);
  int t   = threadIdx.x;

  const float* base = w + ((size_t)e * DOUT_ + ob * 128) * DIN_ + kb * 128;

  v4f vals[16];
  float amax = 0.0f;
  #pragma unroll
  for (int i = 0; i < 16; ++i) {
    int j  = i * 256 + t;
    int o  = j >> 5, kq = j & 31;
    v4f v = __builtin_nontemporal_load((const v4f*)(base + (size_t)o * DIN_ + kq * 4));
    vals[i] = v;
    amax = fmaxf(amax, fmaxf(fmaxf(fabsf(v[0]), fabsf(v[1])),
                             fmaxf(fabsf(v[2]), fabsf(v[3]))));
  }
  red[t] = amax;
  __syncthreads();
  #pragma unroll
  for (int s_ = 128; s_ >= 1; s_ >>= 1) {
    if (t < s_) red[t] = fmaxf(red[t], red[t + s_]);
    __syncthreads();
  }
  float scale = fmaxf(red[0], 1e-12f) * (1.0f / 448.0f);
  float inv   = 1.0f / scale;

  #pragma unroll
  for (int i = 0; i < 16; ++i) {
    int j  = i * 256 + t;
    int o  = j >> 5, kq = j & 31;
    v4f v = vals[i];
    uint32_t d = pack4_e4m3(v[0] * inv, v[1] * inv, v[2] * inv, v[3] * inv);
    int kp  = kq * 4;
    int g   = (kp >> 3) & 1;
    int V   = (kp >> 6) * 8 + ((kp & 63) >> 4) * 2 + ((kp >> 2) & 1);
    int ntl = o >> 4, N = o & 15;
    int L   = N + 16 * g;
    size_t frag = (size_t)(e * NT_ + ob * 8 + ntl) * KB_ + kb;
    wp[(frag * 32 + L) * 16 + V] = d;
  }
  if (t == 0) sw[((size_t)e * OB_ + ob) * KB_ + kb] = scale;
}

__global__ void __launch_bounds__(256) gemm_fp8(const uint32_t* __restrict__ qxp,
                                                const float* __restrict__ sx,
                                                const uint32_t* __restrict__ wp,
                                                const float* __restrict__ sw,
                                                float* __restrict__ out) {
  __shared__ __align__(64) uint8_t smem[2][16][2048];

  const int t    = threadIdx.x;
  const int wave = t >> 5;
  const int lane = t & 31;
  const int g    = lane >> 4;

  const int bid = blockIdx.x;
  const int cg  = bid & 7;
  const int rg  = (bid >> 3) & 15;
  const int e   = bid >> 7;

  const int rtbase  = e * RTE_ + rg * 8;
  const int rowbase = (rtbase + wave) * 16;
  const int nt0     = cg * 8;
  const float* swp  = sw + ((size_t)e * OB_ + cg) * KB_;

  auto issue_async = [&](int kb, int sel) {
    #pragma unroll
    for (int j = 0; j < 8; ++j) {
      int linear = j * 256 + t;
      int f    = linear >> 7;
      int slot = linear & 127;
      const uint32_t* src = (f < 8)
        ? qxp + ((size_t)(rtbase + f) * KB_ + kb) * FRAG_DW + slot * 4
        : wp  + ((size_t)(e * NT_ + nt0 + (f - 8)) * KB_ + kb) * FRAG_DW + slot * 4;
      async_copy_b128(src, &smem[sel][f][slot * 16]);
    }
  };

  v8f acc[8];
  #pragma unroll
  for (int n = 0; n < 8; ++n) acc[n] = (v8f){0.f,0.f,0.f,0.f,0.f,0.f,0.f,0.f};

  issue_async(0, 0);

  for (int kb = 0; kb < KB_; ++kb) {
    const int sel = kb & 1;
    wait_async0();
    __syncthreads();
    if (kb + 1 < KB_) issue_async(kb + 1, sel ^ 1);

    float swv = swp[kb];
    float cs[8];
    #pragma unroll
    for (int i = 0; i < 8; ++i)
      cs[i] = sx[(size_t)(rowbase + i + 8 * g) * KB_ + kb] * swv;

    v16i afrag = *(const v16i*)&smem[sel][wave][lane * 64];

    #pragma unroll
    for (int n = 0; n < 8; ++n) {
      v16i bfrag = *(const v16i*)&smem[sel][8 + n][lane * 64];
      v8f zero = (v8f){0.f,0.f,0.f,0.f,0.f,0.f,0.f,0.f};
      v8f part = __builtin_amdgcn_wmma_f32_16x16x128_fp8_fp8(
          afrag, bfrag, (short)0, zero, false, false);
      #pragma unroll
      for (int i = 0; i < 8; ++i)
        acc[n][i] = fmaf(part[i], cs[i], acc[n][i]);
    }
  }

  const int N = lane & 15;
  #pragma unroll
  for (int n = 0; n < 8; ++n) {
    int col = (nt0 + n) * 16 + N;
    #pragma unroll
    for (int i = 0; i < 8; ++i) {
      int row = rowbase + i + 8 * g;
      __builtin_nontemporal_store(acc[n][i], &out[(size_t)row * DOUT_ + col]);
    }
  }
}

extern "C" void kernel_launch(void* const* d_in, const int* in_sizes, int n_in,
                              void* d_out, int out_size, void* d_ws, size_t ws_size,
                              hipStream_t stream) {
  const float* x = (const float*)d_in[0];
  const float* w = (const float*)d_in[2];
  float* out = (float*)d_out;

  uint8_t* ws = (uint8_t*)d_ws;
  uint32_t* qxp = (uint32_t*)ws;
  uint32_t* wp  = (uint32_t*)(ws + (size_t)SEQ_ * DIN_);
  float*    sx  = (float*)(ws + (size_t)SEQ_ * DIN_ + (size_t)NE_ * DOUT_ * DIN_);
  float*    sw  = sx + (size_t)SEQ_ * KB_;

  quant_act<<<(SEQ_ * KB_) / 8, 256, 0, stream>>>(x, qxp, sx);
  quant_wgt<<<NE_ * OB_ * KB_, 256, 0, stream>>>(w, wp, sw);
  gemm_fp8<<<NE_ * 16 * 8, 256, 0, stream>>>(qxp, sx, wp, sw, out);
}
